// BinaryConv_30511447670909
// MI455X (gfx1250) — 8-bit matrix operand order repaired and checked by reading, compiled, not run on hardware
//
#include <hip/hip_runtime.h>

typedef int  v8i __attribute__((ext_vector_type(8)));
typedef int  v4i __attribute__((ext_vector_type(4)));
typedef int  v2i __attribute__((ext_vector_type(2)));

typedef int gccv4i __attribute__((vector_size(16)));
typedef __attribute__((address_space(1))) gccv4i as1_v4i;
typedef __attribute__((address_space(3))) gccv4i as3_v4i;

#define HH     56
#define WW     56
#define C_IN   256
#define C_OUT  256
#define NBATCH 32
#define SP     (HH * WW)
#define M_TOT  (NBATCH * SP)
#define K_TOT  (C_IN * 9)
#define KCHUNKS (K_TOT / 64)

#if defined(__has_builtin)
#if __has_builtin(__builtin_amdgcn_global_load_async_to_lds_b128)
#define USE_ASYNC_LDS 1
#endif
#endif
#ifndef USE_ASYNC_LDS
#define USE_ASYNC_LDS 0
#endif

#if USE_ASYNC_LDS
#if __has_builtin(__builtin_amdgcn_s_wait_asynccnt)
#define WAIT_ASYNC(n) __builtin_amdgcn_s_wait_asynccnt(n)
#else
#define WAIT_ASYNC(n) asm volatile("s_wait_asynccnt %0" ::"i"(n) : "memory")
#endif
#endif

__global__ void pack_x_sign(const float* __restrict__ x, signed char* __restrict__ xs) {
    int idx = blockIdx.x * 256 + threadIdx.x;
    int c   = idx & 255;
    int spn = idx >> 8;
    int n   = spn / SP;
    int sp  = spn - n * SP;
    float v = x[(n * C_IN + c) * SP + sp];
    xs[idx] = (v > 0.f) ? (signed char)1 : ((v < 0.f) ? (signed char)-1 : (signed char)0);
}

__global__ void pack_w_sign(const float* __restrict__ w, signed char* __restrict__ ws) {
    int idx = blockIdx.x * 256 + threadIdx.x;
    int co  = idx & 255;
    int r   = idx >> 8;
    int ci  = r & 255;
    int tap = r >> 8;
    float v = w[(co * C_IN + ci) * 9 + tap];
    ws[co * K_TOT + (tap << 8) + ci] =
        (v > 0.f) ? (signed char)1 : ((v < 0.f) ? (signed char)-1 : (signed char)0);
}

__device__ __forceinline__ v8i load_a_frag(const signed char* base) {
    const v4i* ap = (const v4i*)base;
    v4i p0 = ap[0];
    v4i p1 = ap[2];
    v8i a;
    a[0] = p0.x; a[1] = p0.y; a[2] = p0.z; a[3] = p0.w;
    a[4] = p1.x; a[5] = p1.y; a[6] = p1.z; a[7] = p1.w;
    return a;
}

__device__ __forceinline__ v8i load_b_frag(const signed char* base) {
    const v4i* bp = (const v4i*)base;
    v4i q0 = bp[0];
    v4i q1 = bp[2];
    v8i b;
    b[0] = q0.x; b[1] = q0.y; b[2] = q0.z; b[3] = q0.w;
    b[4] = q1.x; b[5] = q1.y; b[6] = q1.z; b[7] = q1.w;
    return b;
}

__global__ __launch_bounds__(256, 2)
void binconv_wmma(const signed char* __restrict__ xs,
                  const signed char* __restrict__ wsg,
                  const float* __restrict__ bias,
                  float* __restrict__ out) {
    __shared__ signed char Ald[2][128 * 64];
    __shared__ signed char Bld[2][64 * 64];

    const int tid   = threadIdx.x;
    const int lane  = tid & 31;
    const int hlane = lane & 15;
    const int hisel = lane >> 4;
    const int wave  = tid >> 5;
    const int wm    = wave & 3;
    const int wn    = wave >> 2;

    const int mblk = blockIdx.y * 128;
    const int nblk = blockIdx.x * 64;

    const int arow  = tid >> 1;
    const int abyte = (tid & 1) << 5;
    const int m_st  = mblk + arow;
    const int nb_st = m_st / SP;
    const int sp_st = m_st - nb_st * SP;
    const int oh_st = sp_st / WW;
    const int ow_st = sp_st - oh_st * WW;

    const int bn = tid >> 2;
    const int bk = (tid & 3) << 4;
    const signed char* bsrc0 = wsg + (size_t)(nblk + bn) * K_TOT + bk;

    auto a_inb = [&](int kc_) -> bool {
        int tap = kc_ >> 2;
        int kh  = tap / 3;
        int kw  = tap - kh * 3;
        int ih  = oh_st + kh - 1;
        int iw  = ow_st + kw - 1;
        return ((unsigned)ih < (unsigned)HH) & ((unsigned)iw < (unsigned)WW);
    };
    auto a_src = [&](int kc_) -> const signed char* {
        int tap = kc_ >> 2;
        int c0  = (kc_ & 3) << 6;
        int kh  = tap / 3;
        int kw  = tap - kh * 3;
        int ih  = oh_st + kh - 1;
        int iw  = ow_st + kw - 1;
        int ihc = ih < 0 ? 0 : (ih > HH - 1 ? HH - 1 : ih);
        int iwc = iw < 0 ? 0 : (iw > WW - 1 ? WW - 1 : iw);
        return xs + (((nb_st * HH + ihc) * WW + iwc) << 8) + c0 + abyte;
    };

#if USE_ASYNC_LDS
    auto issue_chunk = [&](int kc_, int buf) {
        const signed char* asrc = a_src(kc_);
        __builtin_amdgcn_global_load_async_to_lds_b128(
            (as1_v4i*)asrc,
            (as3_v4i*)(Ald[buf] + arow * 64 + abyte), 0, 0);
        __builtin_amdgcn_global_load_async_to_lds_b128(
            (as1_v4i*)(asrc + 16),
            (as3_v4i*)(Ald[buf] + arow * 64 + abyte + 16), 0, 0);
        __builtin_amdgcn_global_load_async_to_lds_b128(
            (as1_v4i*)(bsrc0 + (kc_ << 6)),
            (as3_v4i*)(Bld[buf] + bn * 64 + bk), 0, 0);
    };
#endif

    v8i acc[2][2];
#pragma unroll
    for (int mi = 0; mi < 2; ++mi)
#pragma unroll
        for (int ni = 0; ni < 2; ++ni)
            acc[mi][ni] = (v8i){0, 0, 0, 0, 0, 0, 0, 0};

    const int a_base0 = (wm * 32 + hlane) * 64 + hisel * 16;
    const int a_base1 = a_base0 + 16 * 64;
    const int b_base0 = (wn * 32 + hlane) * 64 + hisel * 16;
    const int b_base1 = b_base0 + 16 * 64;

#if USE_ASYNC_LDS
    issue_chunk(0, 0);
#endif

    for (int kc = 0; kc < KCHUNKS; ++kc) {
        const int cur = kc & 1;

#if USE_ASYNC_LDS
        if (kc + 1 < KCHUNKS) {
            issue_chunk(kc + 1, cur ^ 1);
            WAIT_ASYNC(3);
        } else {
            WAIT_ASYNC(0);
        }
        if (!a_inb(kc)) {
            v4i z = {0, 0, 0, 0};
            *(v4i*)(&Ald[cur][arow * 64 + abyte])      = z;
            *(v4i*)(&Ald[cur][arow * 64 + abyte + 16]) = z;
        }
#else
        v4i a0v = {0, 0, 0, 0};
        v4i a1v = {0, 0, 0, 0};
        if (a_inb(kc)) {
            const v4i* ap = (const v4i*)a_src(kc);
            a0v = ap[0];
            a1v = ap[1];
        }
        v4i bv = *(const v4i*)(bsrc0 + (kc << 6));
        if (kc + 1 < KCHUNKS)
            __builtin_prefetch(bsrc0 + ((kc + 1) << 6), 0, 1);
        *(v4i*)(&Ald[cur][arow * 64 + abyte])      = a0v;
        *(v4i*)(&Ald[cur][arow * 64 + abyte + 16]) = a1v;
        *(v4i*)(&Bld[cur][bn * 64 + bk])           = bv;
#endif
        __syncthreads();

        v8i a0 = load_a_frag(Ald[cur] + a_base0);
        v8i a1 = load_a_frag(Ald[cur] + a_base1);
        v8i b0 = load_b_frag(Bld[cur] + b_base0);
        v8i b1 = load_b_frag(Bld[cur] + b_base1);

        acc[0][0] = __builtin_amdgcn_wmma_i32_16x16x64_iu8(true, a0, true, b0, acc[0][0], false, false);
        acc[0][1] = __builtin_amdgcn_wmma_i32_16x16x64_iu8(true, a0, true, b1, acc[0][1], false, false);
        acc[1][0] = __builtin_amdgcn_wmma_i32_16x16x64_iu8(true, a1, true, b0, acc[1][0], false, false);
        acc[1][1] = __builtin_amdgcn_wmma_i32_16x16x64_iu8(true, a1, true, b1, acc[1][1], false, false);
        __syncthreads();
    }

#pragma unroll
    for (int mi = 0; mi < 2; ++mi) {
        const int mbase = mblk + wm * 32 + mi * 16 + hisel * 8;
#pragma unroll
        for (int ni = 0; ni < 2; ++ni) {
            const int   n_ch = nblk + wn * 32 + ni * 16 + hlane;
            const float bv   = bias[n_ch];
#pragma unroll
            for (int r = 0; r < 8; ++r) {
                int   m   = mbase + r;
                int   nb2 = m / SP;
                int   sp2 = m - nb2 * SP;
                float v   = (float)acc[mi][ni][r] + bv;
                out[(nb2 * C_OUT + n_ch) * SP + sp2] =
                    (v > 0.f) ? 1.f : ((v < 0.f) ? -1.f : 0.f);
            }
        }
    }
}

extern "C" void kernel_launch(void* const* d_in, const int* in_sizes, int n_in,
                              void* d_out, int out_size, void* d_ws, size_t ws_size,
                              hipStream_t stream) {
    const float* x    = (const float*)d_in[0];
    const float* wgt  = (const float*)d_in[1];
    const float* bias = (const float*)d_in[2];
    float*       out  = (float*)d_out;

    signed char* xs  = (signed char*)d_ws;
    signed char* wsg = xs + (size_t)M_TOT * C_IN;

    pack_x_sign<<<(M_TOT * C_IN) / 256, 256, 0, stream>>>(x, xs);
    pack_w_sign<<<(9 * C_IN * C_OUT) / 256, 256, 0, stream>>>(wgt, wsg);

    dim3 grid(C_OUT / 64, M_TOT / 128);
    binconv_wmma<<<grid, 256, 0, stream>>>(xs, wsg, bias, out);
}
